// MusicGenerationV2_20761871909358
// MI455X (gfx1250) — compile-verified, matrix operand order checked by reading, not run on hardware
//
#include <hip/hip_runtime.h>

typedef __attribute__((ext_vector_type(16))) _Float16 v16h;
typedef __attribute__((ext_vector_type(8)))  float    v8f;

#define N_NOTES 78
#define IN_DIM  80
#define KPAD    96           // IN_DIM padded to multiple of 32 for f16 WMMA K
#define T_HID   64
#define N_HID   2
#define T_SEQ   128
#define BATCH   32
#define BSPLIT  2            // batch split across blocks
#define MROWS   (BATCH / BSPLIT)   // 16 -> one M tile
#define GATES   (4 * T_HID)        // 256
#define NT      (GATES / 16)       // 16 N tiles
#define KT_IH   (KPAD / 32)        // 3
#define KT_HH   (T_HID / 32)       // 2
#define KT_ALL  (KT_IH + KT_HH)    // 5
#define FRAG_H  512                // halves per 16x32 fragment (32 lanes * 16)

// CDNA5 16-bit fragment lane layout (ISA 7.12.2):
//   lanes 0-15 : row = lane,    K in {0..7, 16..23}
//   lanes 16-31: row = lane-16, K in {8..15, 24..31}
__device__ __forceinline__ int frag_hidx(int row, int kl) {
    int l = (row & 15) + (((kl >> 3) & 1) << 4);
    int q = (kl & 7) | ((kl & 16) >> 1);
    return l * 16 + q;
}

// ---- branchless transcendentals on CDNA5 TRANS units ----
__device__ __forceinline__ float fast_exp(float x) {
#if __has_builtin(__builtin_amdgcn_exp2f)
    return __builtin_amdgcn_exp2f(x * 1.4426950408889634f);   // v_exp_f32
#else
    return __expf(x);
#endif
}
__device__ __forceinline__ float fast_rcp(float x) {
#if __has_builtin(__builtin_amdgcn_rcpf)
    return __builtin_amdgcn_rcpf(x);                          // v_rcp_f32
#else
    return 1.0f / x;
#endif
}
__device__ __forceinline__ float sigm(float x) {
    return fast_rcp(1.0f + fast_exp(-x));
}
__device__ __forceinline__ float fast_tanh(float x) {
#if __has_builtin(__builtin_amdgcn_tanhf)
    return __builtin_amdgcn_tanhf(x);                         // v_tanh_f32
#elif __has_builtin(__builtin_amdgcn_tanh_f32)
    return __builtin_amdgcn_tanh_f32(x);
#else
    float e = fast_exp(-2.0f * x);
    return (1.0f - e) * fast_rcp(1.0f + e);
#endif
}

// ---------------------------------------------------------------------------
// Stage 1: 78 grouped time-LSTMs. One block per (note, batch-half).
// Weight fragments resident in VGPRs across the whole 128-step scan; per step
// the 5 A fragments are loaded up front (pipelined) then 10 WMMAs issue.
// ---------------------------------------------------------------------------
__global__ __launch_bounds__(256)
void time_lstm_kernel(const float* __restrict__ x,      // [B,78,T,80]
                      const float* __restrict__ Wih,    // [78,256,80]
                      const float* __restrict__ Whh,    // [78,256,64]
                      const float* __restrict__ bih,    // [78,256]
                      const float* __restrict__ bhh,    // [78,256]
                      float* __restrict__ hs_t)         // [T][B][78][64]
{
    // LDS layout (bytes):
    //  wih frags : NT*KT_IH*1024 = 49152  (read once into VGPRs)
    //  whh frags : NT*KT_HH*1024 = 32768  (read once into VGPRs)
    //  x   frags : KT_IH*1024    =  3072
    //  h   frags : KT_HH*1024    =  2048
    //  g         : 16*256*4      = 16384
    //  bias      : 256*4         =  1024   -> total 104448
    __shared__ __align__(32) char smem[104448];
    _Float16* s_wih = (_Float16*)(smem);
    _Float16* s_whh = (_Float16*)(smem + 49152);
    _Float16* s_x   = (_Float16*)(smem + 49152 + 32768);
    _Float16* s_h   = (_Float16*)(smem + 49152 + 32768 + 3072);
    float*    s_g   = (float*)   (smem + 49152 + 32768 + 3072 + 2048);
    float*    s_b   = (float*)   (smem + 49152 + 32768 + 3072 + 2048 + 16384);

    const int note  = blockIdx.x / BSPLIT;
    const int bhalf = blockIdx.x % BSPLIT;
    const int tid   = threadIdx.x;
    const int lane  = tid & 31;
    const int wave  = tid >> 5;

    // ---- init: convert weights f32 -> f16 into WMMA B-fragment layout ----
    const float* wih_p = Wih + (size_t)note * GATES * IN_DIM;
    for (int idx = tid; idx < GATES * IN_DIM; idx += 256) {
        int n = idx / IN_DIM, k = idx % IN_DIM;
        s_wih[((n >> 4) * KT_IH + (k >> 5)) * FRAG_H + frag_hidx(n, k & 31)] =
            (_Float16)wih_p[idx];
    }
    for (int idx = tid; idx < GATES * (KPAD - IN_DIM); idx += 256) {
        int n = idx / (KPAD - IN_DIM);
        int k = IN_DIM + idx % (KPAD - IN_DIM);
        s_wih[((n >> 4) * KT_IH + (k >> 5)) * FRAG_H + frag_hidx(n, k & 31)] =
            (_Float16)0.f;
    }
    const float* whh_p = Whh + (size_t)note * GATES * T_HID;
    for (int idx = tid; idx < GATES * T_HID; idx += 256) {
        int n = idx / T_HID, k = idx % T_HID;
        s_whh[((n >> 4) * KT_HH + (k >> 5)) * FRAG_H + frag_hidx(n, k & 31)] =
            (_Float16)whh_p[idx];
    }
    if (tid < GATES) s_b[tid] = bih[note * GATES + tid] + bhh[note * GATES + tid];
    // zero K-pad region of x fragments (only K<80 rewritten per step)
    for (int idx = tid; idx < MROWS * (KPAD - IN_DIM); idx += 256) {
        int m = idx / (KPAD - IN_DIM);
        int k = IN_DIM + idx % (KPAD - IN_DIM);
        s_x[(k >> 5) * FRAG_H + frag_hidx(m, k & 31)] = (_Float16)0.f;
    }
    // zero h fragments (h0 = 0)
    for (int idx = tid; idx < KT_HH * FRAG_H; idx += 256) s_h[idx] = (_Float16)0.f;
    // stage x(t=0)
    for (int idx = tid; idx < MROWS * IN_DIM; idx += 256) {
        int m = idx / IN_DIM, k = idx % IN_DIM;
        int bg = bhalf * MROWS + m;
        const float* xp = &x[(((size_t)bg * N_NOTES + note) * T_SEQ + 0) * IN_DIM + k];
        s_x[(k >> 5) * FRAG_H + frag_hidx(m, k & 31)] = (_Float16)(*xp);
    }

    // cell state registers: thread owns (b_local = tid>>4, h = (tid&15)*4 + j)
    float creg[4] = {0.f, 0.f, 0.f, 0.f};
    const int bl = tid >> 4;          // 0..15
    const int h0 = (tid & 15) * 4;    // 0..60
    const int bg_nl = bhalf * MROWS + bl;

    __syncthreads();

    // ---- preload the wave's 10 B fragments into VGPRs (loop-invariant) ----
    const int nt0 = 2 * wave, nt1 = 2 * wave + 1;
    v16h bw[2 * KT_ALL];
#pragma unroll
    for (int kt = 0; kt < KT_IH; ++kt) {
        bw[2 * kt]     = *(const v16h*)(s_wih + (nt0 * KT_IH + kt) * FRAG_H + lane * 16);
        bw[2 * kt + 1] = *(const v16h*)(s_wih + (nt1 * KT_IH + kt) * FRAG_H + lane * 16);
    }
#pragma unroll
    for (int kt = 0; kt < KT_HH; ++kt) {
        bw[2 * (KT_IH + kt)] =
            *(const v16h*)(s_whh + (nt0 * KT_HH + kt) * FRAG_H + lane * 16);
        bw[2 * (KT_IH + kt) + 1] =
            *(const v16h*)(s_whh + (nt1 * KT_HH + kt) * FRAG_H + lane * 16);
    }
    // bias float4s (loop-invariant)
    float bi4[4], bf4[4], bg4[4], bo4[4];
    *(float4*)bi4 = *(const float4*)(s_b + h0);
    *(float4*)bf4 = *(const float4*)(s_b + 64 + h0);
    *(float4*)bg4 = *(const float4*)(s_b + 128 + h0);
    *(float4*)bo4 = *(const float4*)(s_b + 192 + h0);

    const int mhi  = (lane >> 4) * 8;
    const int ncol = lane & 15;

    for (int t = 0; t < T_SEQ; ++t) {
        // ---- GEMM: g[16 x 256] = x_t * Wih^T + h * Whh^T ----
        {
            // issue ALL A-fragment loads first so they pipeline (distinct regs)
            v16h a[KT_ALL];
#pragma unroll
            for (int kt = 0; kt < KT_IH; ++kt)
                a[kt] = *(const v16h*)(s_x + kt * FRAG_H + lane * 16);
#pragma unroll
            for (int kt = 0; kt < KT_HH; ++kt)
                a[KT_IH + kt] = *(const v16h*)(s_h + kt * FRAG_H + lane * 16);

            v8f acc0 = {}, acc1 = {};
#pragma unroll
            for (int kt = 0; kt < KT_ALL; ++kt) {
                acc0 = __builtin_amdgcn_wmma_f32_16x16x32_f16(
                    false, a[kt], false, bw[2 * kt], (short)0, acc0, false, false);
                acc1 = __builtin_amdgcn_wmma_f32_16x16x32_f16(
                    false, a[kt], false, bw[2 * kt + 1], (short)0, acc1, false, false);
            }
            // C/D layout: VGPR r, lane l -> M = r + 8*(l>>4), N = ntile*16 + (l&15)
#pragma unroll
            for (int r = 0; r < 8; ++r) {
                s_g[(r + mhi) * GATES + nt0 * 16 + ncol] = acc0[r];
                s_g[(r + mhi) * GATES + nt1 * 16 + ncol] = acc1[r];
            }
        }
        __syncthreads();

        // ---- elementwise LSTM cell (gate order i,f,g,o), float4 IO ----
        {
            float gi[4], gf[4], gg[4], go[4], hn[4];
            *(float4*)gi = *(const float4*)(s_g + bl * GATES + h0);
            *(float4*)gf = *(const float4*)(s_g + bl * GATES + 64 + h0);
            *(float4*)gg = *(const float4*)(s_g + bl * GATES + 128 + h0);
            *(float4*)go = *(const float4*)(s_g + bl * GATES + 192 + h0);
#pragma unroll
            for (int j = 0; j < 4; ++j) {
                float cn = sigm(gf[j] + bf4[j]) * creg[j]
                         + sigm(gi[j] + bi4[j]) * fast_tanh(gg[j] + bg4[j]);
                creg[j] = cn;
                hn[j] = sigm(go[j] + bo4[j]) * fast_tanh(cn);
                // new h into A-fragment layout for next step
                int hh = h0 + j;
                s_h[(hh >> 5) * FRAG_H + frag_hidx(bl, hh & 31)] = (_Float16)hn[j];
            }
            // hidden state out: [t][b][note][h], 16B-aligned vector store
            *(float4*)(hs_t + (((size_t)t * BATCH + bg_nl) * N_NOTES + note) * T_HID + h0)
                = *(const float4*)hn;
        }
        // ---- stage x(t+1) in the same phase (s_x free: GEMM t done) ----
        if (t + 1 < T_SEQ) {
            for (int idx = tid; idx < MROWS * IN_DIM; idx += 256) {
                int m = idx / IN_DIM, k = idx % IN_DIM;
                int bg = bhalf * MROWS + m;
                const float* xp =
                    &x[(((size_t)bg * N_NOTES + note) * T_SEQ + (t + 1)) * IN_DIM + k];
                s_x[(k >> 5) * FRAG_H + frag_hidx(m, k & 31)] = (_Float16)(*xp);
                __builtin_prefetch(xp + IN_DIM, 0, 0);  // next timestep's row
            }
        }
        __syncthreads();
    }
}

// ---------------------------------------------------------------------------
// Stage 2: 128 grouped note-LSTMs (64 -> 2), seq over 78 notes. Tiny GEMMs,
// scalar VALU: thread = (batch, gate).
// ---------------------------------------------------------------------------
__global__ __launch_bounds__(256)
void note_lstm_kernel(const float* __restrict__ hs_t,  // [T][B][78][64]
                      const float* __restrict__ Wih,   // [128,8,64]
                      const float* __restrict__ Whh,   // [128,8,2]
                      const float* __restrict__ bih,   // [128,8]
                      const float* __restrict__ bhh,   // [128,8]
                      float* __restrict__ out)         // [B,128,156]
{
    __shared__ float s_w[8 * 64];
    __shared__ float s_wh[8 * 2];
    __shared__ float s_b[8];
    __shared__ float s_h[BATCH * 2];
    __shared__ float s_x[BATCH * 64];
    __shared__ float s_g[BATCH * 8];

    const int t    = blockIdx.x;
    const int tid  = threadIdx.x;
    const int b    = tid >> 3;   // 0..31
    const int gate = tid & 7;    // 0..7 (i0,i1,f0,f1,g0,g1,o0,o1)

    for (int i = tid; i < 8 * 64 / 4; i += 256)
        ((float4*)s_w)[i] = ((const float4*)(Wih + (size_t)t * 8 * 64))[i];
    if (tid < 16) s_wh[tid] = Whh[t * 16 + tid];
    if (tid < 8)  s_b[tid]  = bih[t * 8 + tid] + bhh[t * 8 + tid];
    if (tid < BATCH * 2) s_h[tid] = 0.f;
    float creg = 0.f;
    __syncthreads();

    for (int note = 0; note < N_NOTES; ++note) {
        // stage inputs for this note-step (float4 vectorized)
        for (int idx = tid; idx < BATCH * 64 / 4; idx += 256) {
            int bb = idx >> 4, ii4 = idx & 15;
            ((float4*)s_x)[idx] = *(const float4*)(
                hs_t + (((size_t)t * BATCH + bb) * N_NOTES + note) * T_HID + ii4 * 4);
        }
        __syncthreads();

        float acc = s_b[gate]
                  + s_h[b * 2 + 0] * s_wh[gate * 2 + 0]
                  + s_h[b * 2 + 1] * s_wh[gate * 2 + 1];
#pragma unroll 8
        for (int i = 0; i < 64; ++i) acc += s_x[b * 64 + i] * s_w[gate * 64 + i];
        s_g[b * 8 + gate] = acc;
        __syncthreads();

        if (gate < N_HID) {
            int hh = gate;
            float iv = s_g[b * 8 + hh];
            float fv = s_g[b * 8 + 2 + hh];
            float gv = s_g[b * 8 + 4 + hh];
            float ov = s_g[b * 8 + 6 + hh];
            creg = sigm(fv) * creg + sigm(iv) * fast_tanh(gv);
            float hn = sigm(ov) * fast_tanh(creg);
            s_h[b * 2 + hh] = hn;
            out[((size_t)b * T_SEQ + t) * (N_NOTES * N_HID) + note * 2 + hh] = hn;
        }
        __syncthreads();
    }
}

extern "C" void kernel_launch(void* const* d_in, const int* in_sizes, int n_in,
                              void* d_out, int out_size, void* d_ws, size_t ws_size,
                              hipStream_t stream) {
    (void)in_sizes; (void)n_in; (void)out_size; (void)ws_size;
    const float* x     = (const float*)d_in[0];
    const float* Wt_ih = (const float*)d_in[1];
    const float* Wt_hh = (const float*)d_in[2];
    const float* bt_ih = (const float*)d_in[3];
    const float* bt_hh = (const float*)d_in[4];
    const float* Wn_ih = (const float*)d_in[5];
    const float* Wn_hh = (const float*)d_in[6];
    const float* bn_ih = (const float*)d_in[7];
    const float* bn_hh = (const float*)d_in[8];

    float* hs = (float*)d_ws;   // [T][B][78][64] f32 = 80 MB scratch

    time_lstm_kernel<<<N_NOTES * BSPLIT, 256, 0, stream>>>(
        x, Wt_ih, Wt_hh, bt_ih, bt_hh, hs);
    note_lstm_kernel<<<T_SEQ, 256, 0, stream>>>(
        hs, Wn_ih, Wn_hh, bn_ih, bn_hh, (float*)d_out);
}
